// STSMGNN_64218351010251
// MI455X (gfx1250) — matrix operand order repaired and checked by reading, not compiled, not run on hardware
//
#include <hip/hip_runtime.h>

#define Nn   2048
#define Ee   16384
#define HIDc 128
#define Hh   4
#define Dd   32
#define KH   3

typedef _Float16 h8   __attribute__((ext_vector_type(8)));
typedef _Float16 v16h __attribute__((ext_vector_type(16)));
typedef float    v8f  __attribute__((ext_vector_type(8)));

// ---------------- WMMA helpers (wave32, 16x16x32 f16 -> f32) ----------------

static __device__ __forceinline__ v8f wmma16(v16h a, v16h b, v8f c) {
  // 8 args: (neg_a, A, neg_b, B, c_mod, C, reuse_a, reuse_b)
  return __builtin_amdgcn_wmma_f32_16x16x32_f16(false, a, false, b, (short)0, c,
                                                false, false);
}

// A tile 16x32 f16 from row-major f16 source (base points at [row0][k0], ld in elems).
// lane<16: row=lane, K 0..7 and 16..23 ; lane>=16: row=lane-16, K 8..15 and 24..31
static __device__ __forceinline__ v16h load_a_f16(const _Float16* base, int ld, int lane) {
  const _Float16* p = base + (size_t)(lane & 15) * ld + ((lane >> 4) << 3);
  v16h a;
  *reinterpret_cast<h8*>(&a)       = *reinterpret_cast<const h8*>(p);
  *(reinterpret_cast<h8*>(&a) + 1) = *reinterpret_cast<const h8*>(p + 16);
  return a;
}

// Same A tile but converting from row-major f32 source.
static __device__ __forceinline__ v16h load_a_f32cvt(const float* base, int ld, int lane) {
  const float* p = base + (size_t)(lane & 15) * ld + ((lane >> 4) << 3);
  v16h a;
#pragma unroll
  for (int i = 0; i < 8; ++i) a[i] = (_Float16)p[i];
#pragma unroll
  for (int i = 0; i < 8; ++i) a[8 + i] = (_Float16)p[16 + i];
  return a;
}

// B tile 32x16: B[k][n] = bsrcT[n][k] where bsrcT is row-major f16 (transposed
// operand), base points at [col0][k0]. lane<16: col=lane, K 0..7 and 16..23 ; lane>=16: K 8..15 and 24..31
static __device__ __forceinline__ v16h load_b_f16(const _Float16* baseT, int ld, int lane) {
  const _Float16* p = baseT + (size_t)(lane & 15) * ld + ((lane >> 4) << 3);
  v16h b;
  *reinterpret_cast<h8*>(&b)       = *reinterpret_cast<const h8*>(p);
  *(reinterpret_cast<h8*>(&b) + 1) = *reinterpret_cast<const h8*>(p + 16);
  return b;
}

// ---------------- tiny utility kernels ----------------

__global__ void zero_f32(float* p, int n) {
  int i = blockIdx.x * 256 + threadIdx.x;
  if (i < n) p[i] = 0.f;
}
__global__ void zero_i32(int* p, int n) {
  int i = blockIdx.x * 256 + threadIdx.x;
  if (i < n) p[i] = 0;
}

// WT[c][k] = f16(W[k][c]), 128x128
__global__ void wt_f16(const float* W, _Float16* WT) {
  int i = blockIdx.x * 256 + threadIdx.x;
  int c = i >> 7, k = i & 127;
  WT[c * 128 + k] = (_Float16)W[k * 128 + c];
}

__global__ void deg_count(const int* col, float* deg, int* cnt) {
  int e = blockIdx.x * 256 + threadIdx.x;
  if (e < Ee) {
    atomicAdd(&deg[col[e]], 1.f);
    atomicAdd(&cnt[col[e]], 1);
  }
}

__global__ void deg_inv_k(const float* deg, float* dinv) {
  int i = blockIdx.x * 256 + threadIdx.x;
  if (i < Nn) dinv[i] = deg[i] > 0.f ? rsqrtf(deg[i]) : 0.f;
}

__global__ void norm_b(const int* row, const int* col, const float* dinv,
                       const float* ef, float* normE, float* bvec) {
  int e = blockIdx.x * 256 + threadIdx.x;
  if (e < Ee) {
    float nm = dinv[row[e]] * dinv[col[e]];
    normE[e] = nm;
    atomicAdd(&bvec[col[e]], nm * ef[e]);
  }
}

__global__ void csc_scan(const int* cnt, int* off) {
  if (threadIdx.x == 0 && blockIdx.x == 0) {
    int s = 0;
    for (int i = 0; i < Nn; ++i) { off[i] = s; s += cnt[i]; }
    off[Nn] = s;
  }
}

__global__ void csc_fill(const int* col, const int* off, int* cur, int* edges) {
  int e = blockIdx.x * 256 + threadIdx.x;
  if (e < Ee) {
    int c = col[e];
    int p = atomicAdd(&cur[c], 1);
    edges[off[c] + p] = e;
  }
}

// ---------------- QKV projection: 16x16 tile per wave, K=128 in 4 WMMAs ----------------

__global__ void qkv_kernel(const float* x, const _Float16* WqT, const _Float16* WkT,
                           const _Float16* WvT, const float* bq, const float* bk,
                           const float* bv, _Float16* Qf, _Float16* Kf, float* Vf32) {
  int id = blockIdx.x;
  int ct = id & 7;  id >>= 3;
  int sel = id % 3; int rt = id / 3;
  int lane = threadIdx.x;
  const _Float16* WT = sel == 0 ? WqT : (sel == 1 ? WkT : WvT);
  const float* bias  = sel == 0 ? bq  : (sel == 1 ? bk  : bv);
  int r0 = rt * 16, c0 = ct * 16;
  v8f acc = {};
#pragma unroll
  for (int kk = 0; kk < HIDc; kk += 32) {
    v16h a = load_a_f32cvt(x + (size_t)r0 * HIDc + kk, HIDc, lane);
    v16h b = load_b_f16(WT + (size_t)c0 * HIDc + kk, HIDc, lane);
    acc = wmma16(a, b, acc);
  }
  int cc = c0 + (lane & 15), hi = lane >> 4;
  float bb = bias[cc];
  _Float16* outF = sel == 0 ? Qf : Kf;
#pragma unroll
  for (int j = 0; j < 8; ++j) {
    int rr = r0 + j + 8 * hi;
    float v = acc[j] + bb;
    if (sel == 2) Vf32[(size_t)rr * HIDc + cc] = v;
    else          outF[(size_t)rr * HIDc + cc] = (_Float16)v;
  }
}

// Vt[h][d][n] = f16(V[n][h*32+d])
__global__ void vt_kernel(const float* Vf32, _Float16* Vt) {
  int i = blockIdx.x * 256 + threadIdx.x;  // N*128
  int n = i >> 7, c = i & 127;
  Vt[(size_t)c * Nn + n] = (_Float16)Vf32[(size_t)n * HIDc + c];
}

__global__ void hidden_init(const float* Vf32, const float* hopwise, float* hidden) {
  int i = blockIdx.x * 256 + threadIdx.x;
  if (i < Nn * HIDc) hidden[i] = Vf32[i] * hopwise[0];
}

// ---------------- sim0 = exp(Q K^T / sqrt(32)) : exactly 1 WMMA per 16x16 tile ----------------

__global__ void sim0_kernel(const _Float16* Qf, const _Float16* Kf, float* sim) {
  int id = blockIdx.x;
  int ct = id & 127; id >>= 7;
  int rt = id & 127; id >>= 7;
  int h = id;
  int lane = threadIdx.x;
  v16h a = load_a_f16(Qf + (size_t)(rt * 16) * HIDc + h * Dd, HIDc, lane);
  v16h b = load_b_f16(Kf + (size_t)(ct * 16) * HIDc + h * Dd, HIDc, lane);
  v8f c = {};
  c = wmma16(a, b, c);
  const float scale = 0.1767766952966369f;  // 1/sqrt(32)
  float* dst = sim + ((size_t)h * Nn + rt * 16) * Nn + ct * 16;
  int col = lane & 15, hi = lane >> 4;
#pragma unroll
  for (int j = 0; j < 8; ++j)
    dst[(size_t)(j + 8 * hi) * Nn + col] = __expf(c[j] * scale);
}

// ---------------- hop SpMM (CSC gather, no atomics on sim) + fused row-sum ----------------
// block 256 = 8 waves; block covers 16 dest rows x 256 cols; wave covers 32 cols.
__global__ void spmm_kernel(const float* __restrict__ simPrev, float* __restrict__ simNew,
                            const int* rowI, const float* normE, const float* bvec,
                            const int* off, const int* cscE, float* Crow) {
  int w = threadIdx.x >> 5, lane = threadIdx.x & 31;
  int rt = blockIdx.y, h = blockIdx.z;
  int col = blockIdx.x * 256 + w * 32 + lane;
  const float* sp = simPrev + (size_t)h * Nn * Nn;
  float* sn       = simNew  + (size_t)h * Nn * Nn;
  float acc[16];
#pragma unroll
  for (int j = 0; j < 16; ++j) acc[j] = 0.f;
  for (int j = 0; j < 16; ++j) {
    int r = rt * 16 + j;
    int e0 = off[r], e1 = off[r + 1];
    float a = 0.f;
    for (int t = e0; t < e1; ++t) {
      int e = cscE[t];
      int src = rowI[e];
      if (t + 1 < e1)  // keep next gathered sim row moving toward the WGP
        __builtin_prefetch(&sp[(size_t)rowI[cscE[t + 1]] * Nn + col], 0, 1);
      a += normE[e] * sp[(size_t)src * Nn + col];
    }
    acc[j] = a + bvec[r];
  }
#pragma unroll
  for (int j = 0; j < 16; ++j)
    sn[(size_t)(rt * 16 + j) * Nn + col] = acc[j];
  // fused row-sum (C): reduce this wave's 32 columns, one atomic per row per wave
#pragma unroll
  for (int j = 0; j < 16; ++j) {
    float s = acc[j];
    for (int o = 16; o > 0; o >>= 1) s += __shfl_down(s, o, 32);
    if (lane == 0) atomicAdd(&Crow[h * Nn + rt * 16 + j], s);
  }
}

// ---------------- Hm = sim @ V (WMMA over K=2048 split across 8 waves) fused with
//                  hidden += gamma * Hm / (C + 1e-5) ----------------
__global__ void hm_gemm_kernel(const float* __restrict__ simNew, const _Float16* Vt,
                               const float* Crow, const float* hopwise,
                               const float* headwise, float* hidden, int hop) {
  __shared__ float red[8][16 * 32];
  int w = threadIdx.x >> 5, lane = threadIdx.x & 31;
  int rt = blockIdx.x, h = blockIdx.y;
  const float* sp = simNew + ((size_t)h * Nn + rt * 16) * Nn;
  v8f acc0 = {}, acc1 = {};
  int m0 = w * 256;
#pragma unroll
  for (int mc = 0; mc < 256; mc += 32) {
    v16h a  = load_a_f32cvt(sp + m0 + mc, Nn, lane);
    v16h b0 = load_b_f16(Vt + (size_t)(h * Dd +  0) * Nn + m0 + mc, Nn, lane);
    v16h b1 = load_b_f16(Vt + (size_t)(h * Dd + 16) * Nn + m0 + mc, Nn, lane);
    acc0 = wmma16(a, b0, acc0);
    acc1 = wmma16(a, b1, acc1);
  }
  int rl = 8 * (lane >> 4), cl = lane & 15;
#pragma unroll
  for (int j = 0; j < 8; ++j) {
    red[w][(j + rl) * 32 + cl]      = acc0[j];
    red[w][(j + rl) * 32 + 16 + cl] = acc1[j];
  }
  __syncthreads();
  // layerwise = softmax over heads of headwise[:, hop]; gamma = hopwise[hop+1]*layerwise[h]
  float e0 = __expf(headwise[0 * KH + hop]), e1 = __expf(headwise[1 * KH + hop]);
  float e2 = __expf(headwise[2 * KH + hop]), e3 = __expf(headwise[3 * KH + hop]);
  float gamma = hopwise[hop + 1] * __expf(headwise[h * KH + hop]) / (e0 + e1 + e2 + e3);
  for (int t = threadIdx.x; t < 16 * 32; t += 256) {
    float s = 0.f;
#pragma unroll
    for (int ww = 0; ww < 8; ++ww) s += red[ww][t];
    int r = t >> 5, d = t & 31;
    int n = rt * 16 + r;
    float Cv = Crow[h * Nn + n] + 1e-5f;
    hidden[(size_t)n * HIDc + h * Dd + d] += gamma * s / Cv;
  }
}

// ---------------- out = hidden @ Wo + bo ----------------

__global__ void out_kernel(const float* hidden, const _Float16* WoT, const float* bo,
                           float* out) {
  int ct = blockIdx.x & 7, rt = blockIdx.x >> 3;
  int lane = threadIdx.x;
  int r0 = rt * 16, c0 = ct * 16;
  v8f acc = {};
#pragma unroll
  for (int kk = 0; kk < HIDc; kk += 32) {
    v16h a = load_a_f32cvt(hidden + (size_t)r0 * HIDc + kk, HIDc, lane);
    v16h b = load_b_f16(WoT + (size_t)c0 * HIDc + kk, HIDc, lane);
    acc = wmma16(a, b, acc);
  }
  int cc = c0 + (lane & 15), hi = lane >> 4;
  float bb = bo[cc];
#pragma unroll
  for (int j = 0; j < 8; ++j)
    out[(size_t)(r0 + j + 8 * hi) * HIDc + cc] = acc[j] + bb;
}

// ---------------- host ----------------

extern "C" void kernel_launch(void* const* d_in, const int* in_sizes, int n_in,
                              void* d_out, int out_size, void* d_ws, size_t ws_size,
                              hipStream_t stream) {
  (void)in_sizes; (void)n_in; (void)out_size; (void)ws_size;
  const float* x        = (const float*)d_in[0];
  const float* ef       = (const float*)d_in[1];
  const float* Wq       = (const float*)d_in[2];
  const float* bq       = (const float*)d_in[3];
  const float* Wk       = (const float*)d_in[4];
  const float* bk       = (const float*)d_in[5];
  const float* Wv       = (const float*)d_in[6];
  const float* bv       = (const float*)d_in[7];
  const float* Wo       = (const float*)d_in[8];
  const float* bo       = (const float*)d_in[9];
  const float* hopwise  = (const float*)d_in[10];
  const float* headwise = (const float*)d_in[11];
  const int*   eidx     = (const int*)d_in[12];
  const int* rowI = eidx;
  const int* colI = eidx + Ee;
  float* out = (float*)d_out;

  char* ws = (char*)d_ws;
  size_t off = 0;
  auto alloc = [&](size_t bytes) -> void* {
    void* p = (void*)(ws + off);
    off += bytes;
    off = (off + 255) & ~(size_t)255;
    return p;
  };

  float*    simA  = (float*)alloc(sizeof(float) * Hh * Nn * Nn);   // 64 MB (L2-resident)
  float*    simB  = (float*)alloc(sizeof(float) * Hh * Nn * Nn);   // 64 MB
  _Float16* Qf    = (_Float16*)alloc(2ull * Nn * HIDc);
  _Float16* Kf    = (_Float16*)alloc(2ull * Nn * HIDc);
  float*    Vf32  = (float*)alloc(4ull * Nn * HIDc);
  _Float16* Vt    = (_Float16*)alloc(2ull * Hh * Dd * Nn);
  _Float16* WqT   = (_Float16*)alloc(2ull * HIDc * HIDc);
  _Float16* WkT   = (_Float16*)alloc(2ull * HIDc * HIDc);
  _Float16* WvT   = (_Float16*)alloc(2ull * HIDc * HIDc);
  _Float16* WoT   = (_Float16*)alloc(2ull * HIDc * HIDc);
  float*    deg   = (float*)alloc(4ull * Nn);
  float*    dinv  = (float*)alloc(4ull * Nn);
  float*    normE = (float*)alloc(4ull * Ee);
  float*    bvec  = (float*)alloc(4ull * Nn);
  float*    Crow  = (float*)alloc(4ull * Hh * Nn);
  float*    hidden= (float*)alloc(4ull * Nn * HIDc);
  int*      cnt   = (int*)alloc(4ull * Nn);
  int*      cur   = (int*)alloc(4ull * Nn);
  int*      cscOff= (int*)alloc(4ull * (Nn + 1));
  int*      cscE  = (int*)alloc(4ull * Ee);

  zero_f32<<<(Nn + 255) / 256, 256, 0, stream>>>(deg, Nn);
  zero_f32<<<(Nn + 255) / 256, 256, 0, stream>>>(bvec, Nn);
  zero_i32<<<(Nn + 255) / 256, 256, 0, stream>>>(cnt, Nn);
  zero_i32<<<(Nn + 255) / 256, 256, 0, stream>>>(cur, Nn);

  wt_f16<<<64, 256, 0, stream>>>(Wq, WqT);
  wt_f16<<<64, 256, 0, stream>>>(Wk, WkT);
  wt_f16<<<64, 256, 0, stream>>>(Wv, WvT);
  wt_f16<<<64, 256, 0, stream>>>(Wo, WoT);

  deg_count<<<Ee / 256, 256, 0, stream>>>(colI, deg, cnt);
  deg_inv_k<<<Nn / 256, 256, 0, stream>>>(deg, dinv);
  norm_b<<<Ee / 256, 256, 0, stream>>>(rowI, colI, dinv, ef, normE, bvec);
  csc_scan<<<1, 1, 0, stream>>>(cnt, cscOff);
  csc_fill<<<Ee / 256, 256, 0, stream>>>(colI, cscOff, cur, cscE);

  qkv_kernel<<<128 * 3 * 8, 32, 0, stream>>>(x, WqT, WkT, WvT, bq, bk, bv, Qf, Kf, Vf32);
  vt_kernel<<<Nn * HIDc / 256, 256, 0, stream>>>(Vf32, Vt);
  hidden_init<<<Nn * HIDc / 256, 256, 0, stream>>>(Vf32, hopwise, hidden);
  sim0_kernel<<<Hh * 128 * 128, 32, 0, stream>>>(Qf, Kf, simA);

  float* sp = simA;
  float* sn = simB;
  for (int hop = 0; hop < KH; ++hop) {
    zero_f32<<<(Hh * Nn + 255) / 256, 256, 0, stream>>>(Crow, Hh * Nn);
    spmm_kernel<<<dim3(8, 128, Hh), 256, 0, stream>>>(sp, sn, rowI, normE, bvec,
                                                      cscOff, cscE, Crow);
    hm_gemm_kernel<<<dim3(128, Hh), 256, 0, stream>>>(sn, Vt, Crow, hopwise,
                                                      headwise, hidden, hop);
    float* t = sp; sp = sn; sn = t;
  }

  out_kernel<<<128 * 8, 32, 0, stream>>>(hidden, WoT, bo, out);
}
